// TriangleAttention_64493228917017
// MI455X (gfx1250) — matrix operand order repaired and checked by reading, not compiled, not run on hardware
//
#include <hip/hip_runtime.h>
#include <hip/hip_bf16.h>
#include <math.h>

#define NSEQ   384
#define CDIM   128
#define NHEAD  4
#define DHEAD  32
#define NROWS  (NSEQ * NSEQ)              /* 147456 */
#define QK_SCALE 0.17677669529663687f     /* 1/sqrt(32) */

typedef __attribute__((ext_vector_type(16))) __bf16 v16bf;
typedef __attribute__((ext_vector_type(8)))  __bf16 v8bf;
typedef __attribute__((ext_vector_type(4)))  __bf16 v4bf;
typedef __attribute__((ext_vector_type(8)))  float  v8f;
typedef __attribute__((ext_vector_type(4)))  unsigned int v4u;

union Frag16 {
  v16bf bf;
  v4u q[2];
  unsigned int w[8];
};

static __device__ __forceinline__ v8f vzero8() {
  v8f z = {0.f, 0.f, 0.f, 0.f, 0.f, 0.f, 0.f, 0.f};
  return z;
}

static __device__ __forceinline__ v8f wmma_bf16(v16bf a, v16bf b, v8f c) {
  return __builtin_amdgcn_wmma_f32_16x16x32_bf16(false, a, false, b, (short)0, c,
                                                 false, false);
}

// Async global->LDS copy, 16 bytes per lane. LDS byte address = low 32 bits of
// the generic pointer (LDS aperture maps addr[31:0]); tracked by ASYNCcnt.
static __device__ __forceinline__ void async_copy16(void* lds_dst, const void* gsrc) {
  unsigned ldsaddr = (unsigned)(size_t)lds_dst;
  asm volatile("global_load_async_to_lds_b128 %0, %1, off"
               :: "v"(ldsaddr), "v"(gsrc) : "memory");
}
static __device__ __forceinline__ void async_wait0() {
  asm volatile("s_wait_asynccnt 0" ::: "memory");
}

// A fragment (16x32 bf16) from LDS row-major [m][k].
// Lane L: m = L&15; K groups of 8 per ISA layout, halves offset by (L>>4)*8.
static __device__ __forceinline__ v16bf load_a_frag(const __bf16* lds, int stride,
                                                    int m_base, int k_base, int lane) {
  Frag16 f;
  const __bf16* p =
      lds + (m_base + (lane & 15)) * stride + k_base + ((lane >> 4) << 3);
#pragma unroll
  for (int i = 0; i < 8; ++i) {
    const int k = ((i & 4) << 2) + ((i & 3) << 1);  // 0,2,4,6,16,18,20,22
    f.w[i] = *(const unsigned int*)(p + k);
  }
  return f.bf;
}

// B fragment (32x16 bf16) from LDS stored TRANSPOSED: row-major [n][k].
// Lane L: n = n_base + (L&15); k = k_base + (L>>4)*8 + {0..7, 16..23} -> 2 x ds_load_b128.
static __device__ __forceinline__ v16bf load_bt_frag(const __bf16* lds, int stride,
                                                     int n_base, int k_base, int lane) {
  Frag16 f;
  const __bf16* p =
      lds + (n_base + (lane & 15)) * stride + k_base + ((lane >> 4) << 3);
  f.q[0] = *(const v4u*)p;
  f.q[1] = *(const v4u*)(p + 16);
  return f.bf;
}

// ---------------------------------------------------------------------------
// Kernel 1: LayerNorm -> x (bf16), fused nb_bias[h][q][k] = sum_c x*feat[c][h]
// ---------------------------------------------------------------------------
__global__ __launch_bounds__(256) void ln_nbias_kernel(
    const float* __restrict__ pair_act, const float* __restrict__ ln_scale,
    const float* __restrict__ ln_bias, const float* __restrict__ feat_w,
    __bf16* __restrict__ Xbf, float* __restrict__ nbb) {
  const int lane = threadIdx.x & 31;
  const int wave = threadIdx.x >> 5;
  const int row  = blockIdx.x * 8 + wave;

  const float4 v = *(const float4*)(pair_act + (size_t)row * CDIM + lane * 4);
  float s = v.x + v.y + v.z + v.w;
#pragma unroll
  for (int m = 1; m < 32; m <<= 1) s += __shfl_xor(s, m, 32);
  const float mu = s * (1.0f / CDIM);

  const float d0 = v.x - mu, d1 = v.y - mu, d2 = v.z - mu, d3 = v.w - mu;
  float qv = d0 * d0 + d1 * d1 + d2 * d2 + d3 * d3;
#pragma unroll
  for (int m = 1; m < 32; m <<= 1) qv += __shfl_xor(qv, m, 32);
  const float rstd = rsqrtf(qv * (1.0f / CDIM) + 1e-5f);

  const float4 sc = *(const float4*)(ln_scale + lane * 4);
  const float4 bi = *(const float4*)(ln_bias + lane * 4);
  const float x0 = d0 * rstd * sc.x + bi.x;
  const float x1 = d1 * rstd * sc.y + bi.y;
  const float x2 = d2 * rstd * sc.z + bi.z;
  const float x3 = d3 * rstd * sc.w + bi.w;

  v4bf xb;
  xb[0] = (__bf16)x0; xb[1] = (__bf16)x1; xb[2] = (__bf16)x2; xb[3] = (__bf16)x3;
  *(v4bf*)(Xbf + (size_t)row * CDIM + lane * 4) = xb;

  const float* f = feat_w + lane * 4 * NHEAD;  // feat_w is [C][H]
  float p[NHEAD];
#pragma unroll
  for (int h = 0; h < NHEAD; ++h)
    p[h] = x0 * f[h] + x1 * f[NHEAD + h] + x2 * f[2 * NHEAD + h] + x3 * f[3 * NHEAD + h];
#pragma unroll
  for (int h = 0; h < NHEAD; ++h) {
#pragma unroll
    for (int m = 1; m < 32; m <<= 1) p[h] += __shfl_xor(p[h], m, 32);
  }
  if (lane == 0) {
#pragma unroll
    for (int h = 0; h < NHEAD; ++h) nbb[h * NROWS + row] = p[h];
  }
}

// ---------------------------------------------------------------------------
// Kernel 2: convert weights f32 -> bf16, stored TRANSPOSED [n][k].
// ---------------------------------------------------------------------------
__global__ __launch_bounds__(256) void wconv_kernel(
    const float* __restrict__ qw, const float* __restrict__ kw,
    const float* __restrict__ vw, const float* __restrict__ gw,
    const float* __restrict__ ow, __bf16* __restrict__ WqkvgT,
    __bf16* __restrict__ WoT) {
  const int t = blockIdx.x * 256 + threadIdx.x;  // 0 .. 81919
  if (t < 512 * 128) {
    const int col = t >> 7, c = t & 127;
    const float* src = (col < 128) ? qw : (col < 256) ? kw : (col < 384) ? vw : gw;
    WqkvgT[t] = (__bf16)src[c * 128 + (col & 127)];
  } else {
    const int u = t - 512 * 128;
    if (u < 128 * 128) {
      const int c = u >> 7, hd = u & 127;
      WoT[u] = (__bf16)ow[hd * 128 + c];
    }
  }
}

// ---------------------------------------------------------------------------
// Kernel 3: projection GEMM  P[M][512] = Xbf[M][128] @ W[128][512]
// 8 waves, 128x128 tile per block; async tile staging; B-frags preloaded.
// ---------------------------------------------------------------------------
__global__ __launch_bounds__(256) void proj_gemm_kernel(
    const __bf16* __restrict__ X, const __bf16* __restrict__ WT,
    const float* __restrict__ gating_b, __bf16* __restrict__ P) {
  __shared__ __bf16 lA[128 * 128];    // [m][k]
  __shared__ __bf16 lBt[128 * 128];   // [n][k] (transposed weights)
  const int tid = threadIdx.x, lane = tid & 31, wave = tid >> 5;
  const size_t mbase = (size_t)blockIdx.x * 128;
  const int nbase = blockIdx.y * 128;

  {  // async stage A and B tiles (16B per lane per op)
    const v8bf* srcA = (const v8bf*)(X + mbase * CDIM);
    const v8bf* srcB = (const v8bf*)(WT + (size_t)nbase * 128);
    v8bf* dstA = (v8bf*)lA;
    v8bf* dstB = (v8bf*)lBt;
#pragma unroll
    for (int it = 0; it < 8; ++it) {
      async_copy16(dstA + it * 256 + tid, srcA + it * 256 + tid);
      async_copy16(dstB + it * 256 + tid, srcB + it * 256 + tid);
    }
    async_wait0();
  }
  __syncthreads();

  v8f acc[8];
#pragma unroll
  for (int ns = 0; ns < 8; ++ns) acc[ns] = vzero8();

  const int mrow = wave * 16;
#pragma unroll
  for (int kk = 0; kk < 128; kk += 32) {
    const v16bf a = load_a_frag(lA, 128, mrow, kk, lane);
    v16bf bfr[8];
#pragma unroll
    for (int ns = 0; ns < 8; ++ns) bfr[ns] = load_bt_frag(lBt, 128, ns * 16, kk, lane);
#pragma unroll
    for (int ns = 0; ns < 8; ++ns) acc[ns] = wmma_bf16(a, bfr[ns], acc[ns]);
  }

  const int hi = lane >> 4, nl = lane & 15;
#pragma unroll
  for (int ns = 0; ns < 8; ++ns) {
    const int col = nbase + ns * 16 + nl;
#pragma unroll
    for (int r = 0; r < 8; ++r) {
      const size_t row = mbase + mrow + r + hi * 8;
      float v = acc[ns][r];
      if (col < 128)       v *= QK_SCALE;                               // q
      else if (col >= 384) v = 1.0f / (1.0f + __expf(-(v + gating_b[col - 384])));
      P[row * 512 + col] = (__bf16)v;
    }
  }
}

// ---------------------------------------------------------------------------
// Kernel 4: flash attention. Block = 4 waves = 64 q-rows of one (b,h).
// K staged async; V staged transposed (lVt[d][key]) so V-frags are b128.
// ---------------------------------------------------------------------------
__global__ __launch_bounds__(128) void attn_kernel(
    const __bf16* __restrict__ P, const int* __restrict__ pair_mask,
    const float* __restrict__ nbb, __bf16* __restrict__ WG) {
  __shared__ __bf16 lQ[64 * 32];
  __shared__ __bf16 lK[32 * 32];       // [key][d]  (== B^T layout for K^T)
  __shared__ __bf16 lVt[32 * 32];      // [d][key]  (transposed V)
  __shared__ __bf16 lS[4][16 * 32];    // per-wave P~ tile [q][key]
  const int tid = threadIdx.x, lane = tid & 31, wave = tid >> 5;
  const int qb0 = blockIdx.x * 64;
  const int b = blockIdx.y;
  const int h = blockIdx.z;
  const int qbase = qb0 + wave * 16;
  const int nl = lane & 15, hi = lane >> 4;

  {  // stage Q: 64 rows x 32 cols, async 2x16B per thread
    const int rq = tid >> 1, hc = (tid & 1) * 16;
    const __bf16* src = P + (size_t)(b * NSEQ + qb0 + rq) * 512 + h * DHEAD + hc;
    async_copy16(lQ + rq * 32 + hc, src);
    async_copy16(lQ + rq * 32 + hc + 8, src + 8);
    async_wait0();
  }
  __syncthreads();
  const v16bf aQ = load_a_frag(lQ, 32, wave * 16, 0, lane);

  float mrun[8], lrun[8];
#pragma unroll
  for (int r = 0; r < 8; ++r) { mrun[r] = -INFINITY; lrun[r] = 0.0f; }
  v8f o0 = vzero8(), o1 = vzero8();

  for (int kt = 0; kt < NSEQ; kt += 32) {
    __syncthreads();  // previous step's lK/lVt fully consumed
    {                 // stage K (async) and V^T (register transpose scatter)
      const int rk = tid >> 2, c = (tid & 3) * 8;
      const __bf16* srcK =
          P + (size_t)(b * NSEQ + kt + rk) * 512 + 128 + h * DHEAD + c;
      async_copy16(lK + rk * 32 + c, srcK);
      const v8bf vv = *(const v8bf*)(srcK + 128);
#pragma unroll
      for (int j = 0; j < 8; ++j) lVt[(c + j) * 32 + rk] = vv[j];
      async_wait0();
    }
    __syncthreads();

    const v16bf bK0 = load_bt_frag(lK, 32, 0, 0, lane);
    const v16bf bK1 = load_bt_frag(lK, 32, 16, 0, lane);
    const v16bf bV0 = load_bt_frag(lVt, 32, 0, 0, lane);   // n = d 0..15
    const v16bf bV1 = load_bt_frag(lVt, 32, 16, 0, lane);  // n = d 16..31
    v8f s0 = wmma_bf16(aQ, bK0, vzero8());
    v8f s1 = wmma_bf16(aQ, bK1, vzero8());

    const float mb0 = pair_mask[b * NSEQ + kt + nl]      ? 0.0f : -1e9f;
    const float mb1 = pair_mask[b * NSEQ + kt + 16 + nl] ? 0.0f : -1e9f;
#pragma unroll
    for (int r = 0; r < 8; ++r) {
      const int q = qbase + r + 8 * hi;
      const float* nbq = nbb + (size_t)h * NROWS + (size_t)q * NSEQ + kt + nl;
      s0[r] += mb0 + nbq[0];
      s1[r] += mb1 + nbq[16];
    }

    __bf16* myS = lS[wave];
#pragma unroll
    for (int r = 0; r < 8; ++r) {
      float t = fmaxf(s0[r], s1[r]);
#pragma unroll
      for (int m = 1; m < 16; m <<= 1) t = fmaxf(t, __shfl_xor(t, m, 32));
      const float mnew = fmaxf(mrun[r], t);
      const float corr = __expf(mrun[r] - mnew);
      const float p0 = __expf(s0[r] - mnew);
      const float p1 = __expf(s1[r] - mnew);
      float rs = p0 + p1;
#pragma unroll
      for (int m = 1; m < 16; m <<= 1) rs += __shfl_xor(rs, m, 32);
      lrun[r] = lrun[r] * corr + rs;
      mrun[r] = mnew;
      o0[r] *= corr;
      o1[r] *= corr;
      myS[(r + 8 * hi) * 32 + nl]      = (__bf16)p0;
      myS[(r + 8 * hi) * 32 + 16 + nl] = (__bf16)p1;
    }
    // same-wave DS ordering: our stores are visible to our own loads
    const v16bf aP = load_a_frag(myS, 32, 0, 0, lane);
    o0 = wmma_bf16(aP, bV0, o0);
    o1 = wmma_bf16(aP, bV1, o1);
  }

  // epilogue: normalize, gate, store bf16
#pragma unroll
  for (int r = 0; r < 8; ++r) {
    const int q = qbase + r + 8 * hi;
    const float inv = (lrun[r] > 0.0f) ? 1.0f / lrun[r] : 0.0f;
    const size_t grow = (size_t)(b * NSEQ + q) * 512 + 384 + h * DHEAD;
    const float g0 = (float)P[grow + nl];
    const float g1 = (float)P[grow + 16 + nl];
    const size_t wrow = (size_t)(b * NSEQ + q) * CDIM + h * DHEAD;
    WG[wrow + nl]      = (__bf16)(o0[r] * inv * g0);
    WG[wrow + 16 + nl] = (__bf16)(o1[r] * inv * g1);
  }
}

// ---------------------------------------------------------------------------
// Kernel 5: output GEMM  out[M][128] = WG[M][128] @ o_w[128][128] + o_b (f32)
// ---------------------------------------------------------------------------
__global__ __launch_bounds__(256) void out_gemm_kernel(
    const __bf16* __restrict__ A, const __bf16* __restrict__ WoT,
    const float* __restrict__ o_b, float* __restrict__ out) {
  __shared__ __bf16 lA[128 * 128];
  __shared__ __bf16 lBt[128 * 128];
  const int tid = threadIdx.x, lane = tid & 31, wave = tid >> 5;
  const size_t mbase = (size_t)blockIdx.x * 128;

  {
    const v8bf* srcA = (const v8bf*)(A + mbase * CDIM);
    const v8bf* srcB = (const v8bf*)WoT;
    v8bf* dstA = (v8bf*)lA;
    v8bf* dstB = (v8bf*)lBt;
#pragma unroll
    for (int it = 0; it < 8; ++it) {
      async_copy16(dstA + it * 256 + tid, srcA + it * 256 + tid);
      async_copy16(dstB + it * 256 + tid, srcB + it * 256 + tid);
    }
    async_wait0();
  }
  __syncthreads();

  v8f acc[8];
#pragma unroll
  for (int ns = 0; ns < 8; ++ns) acc[ns] = vzero8();

  const int mrow = wave * 16;
#pragma unroll
  for (int kk = 0; kk < 128; kk += 32) {
    const v16bf a = load_a_frag(lA, 128, mrow, kk, lane);
    v16bf bfr[8];
#pragma unroll
    for (int ns = 0; ns < 8; ++ns) bfr[ns] = load_bt_frag(lBt, 128, ns * 16, kk, lane);
#pragma unroll
    for (int ns = 0; ns < 8; ++ns) acc[ns] = wmma_bf16(a, bfr[ns], acc[ns]);
  }

  const int hi = lane >> 4, nl = lane & 15;
#pragma unroll
  for (int ns = 0; ns < 8; ++ns) {
    const int col = ns * 16 + nl;
#pragma unroll
    for (int r = 0; r < 8; ++r) {
      const size_t row = mbase + mrow + r + hi * 8;
      out[row * CDIM + col] = acc[ns][r] + o_b[col];
    }
  }
}

// ---------------------------------------------------------------------------
extern "C" void kernel_launch(void* const* d_in, const int* in_sizes, int n_in,
                              void* d_out, int out_size, void* d_ws, size_t ws_size,
                              hipStream_t stream) {
  (void)in_sizes; (void)n_in; (void)out_size; (void)ws_size;
  const float* pair_act  = (const float*)d_in[0];
  const int*   pair_mask = (const int*)d_in[1];
  const float* ln_scale  = (const float*)d_in[2];
  const float* ln_bias   = (const float*)d_in[3];
  const float* feat_w    = (const float*)d_in[4];
  const float* q_w       = (const float*)d_in[5];
  const float* k_w       = (const float*)d_in[6];
  const float* v_w       = (const float*)d_in[7];
  const float* gating_w  = (const float*)d_in[8];
  const float* gating_b  = (const float*)d_in[9];
  const float* o_w       = (const float*)d_in[10];
  const float* o_b       = (const float*)d_in[11];
  float* out = (float*)d_out;

  char* ws = (char*)d_ws;
  size_t off = 0;
  auto take = [&](size_t bytes) -> void* {
    void* p = ws + off;
    off += (bytes + 255) & ~(size_t)255;
    return p;
  };
  __bf16* Xbf    = (__bf16*)take((size_t)NROWS * CDIM * 2);   // LN(x) bf16
  float*  nbb    = (float*) take((size_t)NHEAD * NROWS * 4);  // nb_bias [h][q][k]
  __bf16* WqkvgT = (__bf16*)take((size_t)512 * CDIM * 2);     // W^T bf16
  __bf16* WoT    = (__bf16*)take((size_t)CDIM * CDIM * 2);    // o_w^T bf16
  __bf16* P      = (__bf16*)take((size_t)NROWS * 512 * 2);    // q|k|v|gate
  __bf16* WG     = (__bf16*)take((size_t)NROWS * CDIM * 2);   // gated wavg

  ln_nbias_kernel<<<NROWS / 8, 256, 0, stream>>>(pair_act, ln_scale, ln_bias,
                                                 feat_w, Xbf, nbb);
  wconv_kernel<<<(512 * 128 + 128 * 128) / 256, 256, 0, stream>>>(
      q_w, k_w, v_w, gating_w, o_w, WqkvgT, WoT);
  proj_gemm_kernel<<<dim3(NROWS / 128, 4), 256, 0, stream>>>(Xbf, WqkvgT,
                                                             gating_b, P);
  attn_kernel<<<dim3(NSEQ / 64, NSEQ, NHEAD), 128, 0, stream>>>(P, pair_mask,
                                                                nbb, WG);
  out_gemm_kernel<<<NROWS / 128, 256, 0, stream>>>(WG, WoT, o_b, out);
}
